// AdditiveAttention_18356690223188
// MI455X (gfx1250) — matrix operand order repaired and checked by reading, not compiled, not run on hardware
//
#include <hip/hip_runtime.h>
#include <hip/hip_bf16.h>

// B=16, NQ=64, NK=512, QS=KS=H=VD=256
#define BATCH 16
#define NQ    64
#define NK    512
#define HDIM  256
#define VDIM  256

typedef __attribute__((ext_vector_type(16))) __bf16 v16bf;
typedef __attribute__((ext_vector_type(8)))  __bf16 v8bf;
typedef __attribute__((ext_vector_type(4)))  __bf16 v4bf;
typedef __attribute__((ext_vector_type(2)))  __bf16 v2bf;
typedef __attribute__((ext_vector_type(8)))  float  v8f;
typedef __attribute__((ext_vector_type(4)))  float  v4f;
typedef __attribute__((ext_vector_type(2)))  float  v2f;

// ---------------------------------------------------------------------------
// tanh(x) = 1 - 2/(exp2(2*log2e*x)+1): v_exp_f32 + v_rcp_f32 (TRANS path),
// exact at 0, saturates correctly at +/-inf.
// ---------------------------------------------------------------------------
__device__ __forceinline__ float fast_tanh(float x) {
    float t = __builtin_amdgcn_exp2f(x * 2.885390081777927f); // 2*log2(e)
    return 1.0f - 2.0f * __builtin_amdgcn_rcpf(t + 1.0f);
}

// ---------------------------------------------------------------------------
// Batched GEMM: C[b] (MxN f32) = A[b] (MxK f32) * B[b] (KxN f32), bf16 WMMA
// with f32 accumulation. Block tile 32(M) x 128(N), K-step 32.
// 256 threads = 8 waves in a 2x4 grid; each wave owns two 16x16 C tiles
// (shared A fragment -> 2 v_wmma per barrier pair).
// All dims are multiples of the tiles (no bounds checks).
// ---------------------------------------------------------------------------
#define TM 32
#define TN 128
#define TK 32
#define AST 40   // LDS row stride in bf16: 80B rows -> 16B-aligned fragments
#define BST 40

__global__ __launch_bounds__(256)
void gemm_bf16_wmma(const float* __restrict__ A, const float* __restrict__ B,
                    float* __restrict__ C, int N, int K,
                    long long sA, long long sB, long long sC)
{
    __shared__ __align__(16) __bf16 At[TM * AST];   // [m][k]
    __shared__ __align__(16) __bf16 Bt[TN * BST];   // transposed [n][k]

    const int tid  = threadIdx.x;
    const int bz   = blockIdx.z;
    A += (long long)bz * sA;
    B += (long long)bz * sB;
    C += (long long)bz * sC;

    const int m0   = blockIdx.x * TM;
    const int n0   = blockIdx.y * TN;
    const int wave = tid >> 5;
    const int lane = tid & 31;
    const int wm   = wave >> 2;          // 0..1
    const int wn   = wave & 3;           // 0..3

    // --- per-thread staging coordinates (fixed across K loop) ---
    // A: 32x32 tile = 256 float4; thread -> (m = tid/8, k4 = (tid%8)*4)
    const int sa_m  = tid >> 3;
    const int sa_k  = (tid & 7) * 4;
    const float* aSrc = A + (size_t)(m0 + sa_m) * K + sa_k;
    __bf16* aDst = &At[sa_m * AST + sa_k];

    // B: 32x128 tile = 512 k-pair groups of 4 n; thread does 2 groups.
    // group g: n = (g%32)*4, k = (g/32)*2
    const int g0_n = (tid & 31) * 4;
    const int g0_k = (tid >> 5) * 2;        // 0..14
    const int g1_k = g0_k + 16;             // second group: +256 in g
    const float* bSrc0 = B + (size_t)g0_k * N + n0 + g0_n;
    const float* bSrc1 = B + (size_t)g1_k * N + n0 + g0_n;

    // --- per-thread WMMA fragment LDS addresses (fixed) ---
    // A 16x32 bf16: lane = row M=lane%16; K run [16*(lane/16),+16) contiguous.
    // B 32x16 bf16: lane = col N=lane%16; K run [16*(lane/16),+16) contiguous.
    const __bf16* ap  = &At[(wm * 16 + (lane & 15)) * AST + (lane >> 4) * 16];
    const __bf16* bp0 = &Bt[(wn * 32 + (lane & 15)) * BST + (lane >> 4) * 16];
    const __bf16* bp1 = bp0 + 16 * BST;

    v8f acc0 = {};
    v8f acc1 = {};

    for (int k0 = 0; k0 < K; k0 += TK) {
        // ---- load global (all b128, in flight together) ----
        v4f av  = *(const v4f*)(aSrc + k0);
        v4f b00 = *(const v4f*)(bSrc0 + (size_t)k0 * N);
        v4f b01 = *(const v4f*)(bSrc0 + (size_t)(k0 + 1) * N);
        v4f b10 = *(const v4f*)(bSrc1 + (size_t)k0 * N);
        v4f b11 = *(const v4f*)(bSrc1 + (size_t)(k0 + 1) * N);
        __syncthreads();   // previous iteration's fragment reads done

        // ---- convert + store to LDS ----
        v4bf ac;
        #pragma unroll
        for (int j = 0; j < 4; ++j) ac[j] = (__bf16)av[j];
        *(v4bf*)aDst = ac;
        #pragma unroll
        for (int j = 0; j < 4; ++j) {
            v2bf p0; p0[0] = (__bf16)b00[j]; p0[1] = (__bf16)b01[j];
            *(v2bf*)&Bt[(g0_n + j) * BST + g0_k] = p0;
            v2bf p1; p1[0] = (__bf16)b10[j]; p1[1] = (__bf16)b11[j];
            *(v2bf*)&Bt[(g0_n + j) * BST + g1_k] = p1;
        }
        __syncthreads();

        // ---- fragments + 2 WMMAs (shared A) ----
        v8bf a_lo = *(const v8bf*)(ap);
        v8bf a_hi = *(const v8bf*)(ap + 8);
        v16bf afrag = __builtin_shufflevector(a_lo, a_hi,
                        0,1,2,3,4,5,6,7,8,9,10,11,12,13,14,15);
        v8bf c0 = *(const v8bf*)(bp0);
        v8bf c1 = *(const v8bf*)(bp0 + 8);
        v16bf bfrag0 = __builtin_shufflevector(c0, c1,
                        0,1,2,3,4,5,6,7,8,9,10,11,12,13,14,15);
        v8bf d0 = *(const v8bf*)(bp1);
        v8bf d1 = *(const v8bf*)(bp1 + 8);
        v16bf bfrag1 = __builtin_shufflevector(d0, d1,
                        0,1,2,3,4,5,6,7,8,9,10,11,12,13,14,15);

        acc0 = __builtin_amdgcn_wmma_f32_16x16x32_bf16(
                   false, afrag, false, bfrag0, (short)0, acc0, false, false);
        acc1 = __builtin_amdgcn_wmma_f32_16x16x32_bf16(
                   false, afrag, false, bfrag1, (short)0, acc1, false, false);
    }

    // C/D f32 layout: lane = col N=lane%16; VGPR r -> M = r + 8*(lane/16)
    const int rowBase = m0 + wm * 16 + 8 * (lane >> 4);
    const int col0    = n0 + wn * 32 + (lane & 15);
    float* cBase = C + (size_t)rowBase * N + col0;
    #pragma unroll
    for (int r = 0; r < 8; ++r) {
        cBase[(size_t)r * N]      = acc0[r];
        cBase[(size_t)r * N + 16] = acc1[r];
    }
}

// ---------------------------------------------------------------------------
// Scores: scores[b,q,k] = sum_h w_v[h]*tanh(qp[b,q,h]+kp[b,k,h]).
// Block = 16 queries x one batch; kproj streamed in 32-key LDS tiles.
// Thread (qi = tid&15, kg = tid>>4) computes keys {2kg, 2kg+1} per tile.
// All inner-loop LDS traffic is b128 (4 loads serve 8 tanh+FMA elements).
// Strides padded to 260 (== 4 mod 64) -> disjoint banks across rows.
// ---------------------------------------------------------------------------
#define QPST 260

__global__ __launch_bounds__(256)
void addattn_scores(const float* __restrict__ qproj,
                    const float* __restrict__ kproj,
                    const float* __restrict__ wv,
                    float* __restrict__ scores)
{
    __shared__ __align__(16) float qp_s[16 * QPST];
    __shared__ __align__(16) float kp_s[32 * QPST];
    __shared__ __align__(16) float wv_s[HDIM];

    const int tid = threadIdx.x;
    const int b   = blockIdx.y;
    const int q0  = blockIdx.x * 16;

    // stage w_v and the 16-query qproj tile (b128 loads)
    wv_s[tid] = wv[tid];
    #pragma unroll
    for (int i = 0; i < 4; ++i) {
        int idx = tid + 256 * i;              // 1024 float4 groups / 4 = wrong? 16*64=1024 groups
        int qi = idx >> 6, f = idx & 63;      // 64 float4 per 256-wide row
        *(v4f*)&qp_s[qi * QPST + 4 * f] =
            *(const v4f*)&qproj[(size_t)(b * NQ + q0 + qi) * HDIM + 4 * f];
    }

    const int qi = tid & 15;
    const int kg = tid >> 4;                  // 0..15 -> keys 2kg, 2kg+1
    const float* qrow  = &qp_s[qi * QPST];
    const float* krow0 = &kp_s[(2 * kg) * QPST];
    const float* krow1 = &kp_s[(2 * kg + 1) * QPST];

    for (int kt = 0; kt < NK / 32; ++kt) {
        __syncthreads();
        // stage 32-key kproj tile: 32*256 floats = 2048 float4, 8 per thread
        #pragma unroll
        for (int i = 0; i < 8; ++i) {
            int idx = tid + 256 * i;
            int kk = idx >> 6, f = idx & 63;
            *(v4f*)&kp_s[kk * QPST + 4 * f] =
                *(const v4f*)&kproj[(size_t)(b * NK + kt * 32 + kk) * HDIM + 4 * f];
        }
        __syncthreads();

        float acc0 = 0.0f, acc1 = 0.0f;
        #pragma unroll 2
        for (int h = 0; h < HDIM; h += 4) {
            v4f qv  = *(const v4f*)(qrow + h);
            v4f k0v = *(const v4f*)(krow0 + h);
            v4f k1v = *(const v4f*)(krow1 + h);
            v4f wvv = *(const v4f*)(wv_s + h);
            #pragma unroll
            for (int j = 0; j < 4; ++j) {
                acc0 += wvv[j] * fast_tanh(qv[j] + k0v[j]);
                acc1 += wvv[j] * fast_tanh(qv[j] + k1v[j]);
            }
        }
        v2f sc; sc[0] = acc0; sc[1] = acc1;
        *(v2f*)&scores[(size_t)(b * NQ + q0 + qi) * NK + kt * 32 + 2 * kg] = sc;
    }
}

// ---------------------------------------------------------------------------
// Masked softmax over keys, in place. One wave per query row (8 rows/block),
// lane owns 4 contiguous keys per chunk -> all global traffic is b128.
// ---------------------------------------------------------------------------
__global__ __launch_bounds__(256)
void masked_softmax(float* __restrict__ attn, const int* __restrict__ vlens)
{
    const int b    = blockIdx.y;
    const int wave = threadIdx.x >> 5;
    const int lane = threadIdx.x & 31;
    const int q    = blockIdx.x * 8 + wave;
    const int vlen = vlens[b];

    float* row = attn + (size_t)(b * NQ + q) * NK;

    v4f sv[NK / 128];
    float mx = -3.0e38f;
    #pragma unroll
    for (int i = 0; i < NK / 128; ++i) {
        int k = 4 * (lane + 32 * i);
        v4f v = *(const v4f*)(row + k);
        #pragma unroll
        for (int j = 0; j < 4; ++j) {
            float x = (k + j < vlen) ? v[j] : -1.0e6f; // reference uses -1e6
            v[j] = x;
            mx = fmaxf(mx, x);
        }
        sv[i] = v;
    }
    #pragma unroll
    for (int off = 16; off >= 1; off >>= 1)
        mx = fmaxf(mx, __shfl_xor(mx, off, 32));

    float sum = 0.0f;
    #pragma unroll
    for (int i = 0; i < NK / 128; ++i) {
        #pragma unroll
        for (int j = 0; j < 4; ++j) {
            float e = __builtin_amdgcn_exp2f((sv[i][j] - mx) * 1.4426950408889634f);
            sv[i][j] = e;
            sum += e;
        }
    }
    #pragma unroll
    for (int off = 16; off >= 1; off >>= 1)
        sum += __shfl_xor(sum, off, 32);

    float inv = __builtin_amdgcn_rcpf(sum);
    #pragma unroll
    for (int i = 0; i < NK / 128; ++i) {
        v4f v = sv[i];
        #pragma unroll
        for (int j = 0; j < 4; ++j) v[j] *= inv;
        *(v4f*)(row + 4 * (lane + 32 * i)) = v;
    }
}

// ---------------------------------------------------------------------------
// Launch: qproj = Q*Wq ; kproj = K*Wk ; scores ; softmax ; out = attn*V
// Workspace (f32): qproj 1 MB | kproj 8 MB | attn 2 MB
// ---------------------------------------------------------------------------
extern "C" void kernel_launch(void* const* d_in, const int* in_sizes, int n_in,
                              void* d_out, int out_size, void* d_ws, size_t ws_size,
                              hipStream_t stream)
{
    (void)in_sizes; (void)n_in; (void)out_size; (void)ws_size;

    const float* queries = (const float*)d_in[0];
    const float* keys    = (const float*)d_in[1];
    const float* values  = (const float*)d_in[2];
    const int*   vlens   = (const int*)  d_in[3];
    const float* Wq      = (const float*)d_in[4];
    const float* Wk      = (const float*)d_in[5];
    const float* wv      = (const float*)d_in[6];
    float* out = (float*)d_out;

    float* ws    = (float*)d_ws;
    float* qproj = ws;                                        // [B*NQ, H]
    float* kproj = qproj + (size_t)BATCH * NQ * HDIM;         // [B*NK, H]
    float* attn  = kproj + (size_t)BATCH * NK * HDIM;         // [B, NQ, NK]

    // 1) q projection: (B*NQ=1024) x 256, K=256
    gemm_bf16_wmma<<<dim3((BATCH * NQ) / TM, HDIM / TN, 1), 256, 0, stream>>>(
        queries, Wq, qproj, HDIM, 256, 0, 0, 0);

    // 2) k projection: (B*NK=8192) x 256, K=256
    gemm_bf16_wmma<<<dim3((BATCH * NK) / TM, HDIM / TN, 1), 256, 0, stream>>>(
        keys, Wk, kproj, HDIM, 256, 0, 0, 0);

    // 3) additive scores, then masked softmax in place
    addattn_scores<<<dim3(NQ / 16, BATCH), 256, 0, stream>>>(qproj, kproj, wv, attn);
    masked_softmax<<<dim3(NQ / 8, BATCH), 256, 0, stream>>>(attn, vlens);

    // 4) out[b] = attn[b] (64x512) * values[b] (512x256), batched over z
    gemm_bf16_wmma<<<dim3(NQ / TM, VDIM / TN, BATCH), 256, 0, stream>>>(
        attn, values, out, VDIM, NK,
        (long long)NQ * NK, (long long)NK * VDIM, (long long)NQ * VDIM);
}
